// Conv2dTo3d_84645215469866
// MI455X (gfx1250) — matrix operand order repaired and checked by reading, not compiled, not run on hardware
//
#include <hip/hip_runtime.h>
#include <hip/hip_bf16.h>

typedef __attribute__((ext_vector_type(16))) _Float16     v16h;
typedef __attribute__((ext_vector_type(8)))  float        v8f;
typedef __attribute__((ext_vector_type(4)))  unsigned int v4u;
typedef __attribute__((ext_vector_type(8)))  int          v8i;
typedef __attribute__((ext_vector_type(4)))  int          v4i;

// Problem constants (from reference)
#define IMG_H   2048
#define IMG_W   2048
#define BITS    16
#define KD      3
#define KH      5
#define KW      5
#define D_OUT   14              // BITS - KD + 1
#define H_OUT   2044            // IMG_H - KH + 1
#define W_OUT   2044

#define REC_B   32              // bytes per pixel record (16 x f16 bit-planes)
#define ROW_B   (IMG_W * REC_B) // 65536 bytes per image row in V

#define NTILES    4             // j-tiles (16 cols each) per wave
#define STRIP_PX  (NTILES * 16 + 4)    // 68 pixels staged per row (64 + halo)
#define TILE_ROWS 12                   // 8 output rows + 4 halo
#define STRIP_ROW_B (STRIP_PX * REC_B) // 2176 bytes per staged row
#define STRIP_ROW_H (STRIP_PX * 16)    // 1088 halves per staged row

#define AFRAG_BYTES (13 * 32 * 32)     // 13 K-steps x 32 lanes x 32 B = 13312
#define V_BYTES ((size_t)IMG_H * IMG_W * REC_B)   // 128 MiB

// ---------------------------------------------------------------------------
// Phase 1: expand int32 image into z-interleaved f16 bit-plane volume.
// V[(row*IMG_W + col)*16 + z] = (f16)((x>>z)&1); 32 B per pixel.
// No spatial padding needed: TDM out-of-bounds reads return zeros.
// ---------------------------------------------------------------------------
__global__ void bitexpand_kernel(const int* __restrict__ x,
                                 _Float16* __restrict__ V) {
    int idx = blockIdx.x * blockDim.x + threadIdx.x;
    if (idx >= IMG_H * IMG_W) return;
    unsigned int xv = (unsigned int)x[idx];

    unsigned int packed[8];
#pragma unroll
    for (int k = 0; k < 8; ++k) {
        unsigned int b0 = (xv >> (2 * k)) & 1u;
        unsigned int b1 = (xv >> (2 * k + 1)) & 1u;
        packed[k] = (b0 ? 0x3C00u : 0u) | (b1 ? 0x3C000000u : 0u);  // f16 1.0
    }
    uint4* rec = (uint4*)(V + (size_t)idx * 16);
    rec[0] = make_uint4(packed[0], packed[1], packed[2], packed[3]);
    rec[1] = make_uint4(packed[4], packed[5], packed[6], packed[7]);
}

// ---------------------------------------------------------------------------
// Phase 1b: build the constant WMMA A-operand (weight) table once.
// Entry (t, lane): the 16-half fragment lane `lane` holds for K-step t, in
// contiguous k order (k = 16*(lane/16) + elem, B uses the same), not AMD's table order:
//   lanes 0-15 : elems 0-15 = (tap 2t,   z 0..15)
//   lanes 16-31: elems 0-15 = (tap 2t+1, z 0..15)
// A[d, tap*16+z] = w[z-d, tap] for z-d in [0,3), else 0.
// ---------------------------------------------------------------------------
__global__ void build_afrag_kernel(const float* __restrict__ w,   // [3][25]
                                   v16h* __restrict__ Atab) {     // [13*32]
    int idx = blockIdx.x * blockDim.x + threadIdx.x;
    if (idx >= 13 * 32) return;
    int t  = idx >> 5;
    int ln = idx & 31;
    int d    = ln & 15;
    int zoff = (ln >> 4) * 8;
    v16h frag;
#pragma unroll
    for (int e = 0; e < 16; ++e) {
        int  tap = 2 * t + (ln >> 4);
        int  z   = e;
        int  dz  = z - d;
        bool ok  = (tap < 25) & (dz >= 0) & (dz < KD);
        frag[e] = (_Float16)(ok ? w[dz * 25 + tap] : 0.0f);
    }
    Atab[idx] = frag;
}

// ---------------------------------------------------------------------------
// Phase 2: GEMM-formulated 3D conv on WMMA, everything staged via TDM.
//   Y[d, (i,j)] = sum_{tap,z} Awt[d, tap*16+z] * Bbits[tap*16+z, (i,j)]
// Block: 8 waves x 8 output rows x 64 output cols.
//   TDM #1: A-fragment table (13312 B) -> LDS
//   TDM #2: 12 x 68-pixel input patch (26112 B) -> LDS
// Each wave: 4 j-tiles, 13 K-steps, A hoisted across tiles -> 52 WMMA.
// ---------------------------------------------------------------------------
__global__ void __launch_bounds__(256)
conv_wmma_kernel(const _Float16* __restrict__ V,
                 const v16h* __restrict__ Atab,   // [13*32] weight fragments
                 const float* __restrict__ bias,  // [1]
                 float* __restrict__ out) {
    __shared__ v16h Alds[13][32];                           // 13312 B
    __shared__ __align__(32) _Float16 Vtile[TILE_ROWS * STRIP_ROW_H]; // 26112 B

    const int tid  = threadIdx.x;
    const int lane = tid & 31;
    const int wave = tid >> 5;

    const int j0 = blockIdx.x * (NTILES * 16); // strip start: 32 strips x 64
    const int i0 = blockIdx.y * 8;             // block's first output row

    // -------- TDM: stage A table + input patch into LDS --------------------
    if (wave == 0) {
        v4i gz4 = { 0, 0, 0, 0 };
        v8i gz8 = { 0, 0, 0, 0, 0, 0, 0, 0 };

        // ---- descriptor 1: A-fragment table (1 x 13312 B) ----
        {
            unsigned lds_off = (unsigned)(unsigned long long)(void*)&Alds[0][0];
            unsigned long long ga = (unsigned long long)(const void*)Atab;
            v4u g0 = { 1u, lds_off,
                       (unsigned)(ga & 0xFFFFFFFFull),
                       (unsigned)((ga >> 32) & 0x1FFFFFFull) | (2u << 30) };
            v8i g1 = { 0,
                       (int)((AFRAG_BYTES & 0xFFFFu) << 16),       // dim0 lo
                       (int)(((AFRAG_BYTES >> 16) & 0xFFFFu) | (1u << 16)), // dim0 hi | dim1=1
                       (int)((AFRAG_BYTES & 0xFFFFu) << 16),       // tile0 = 13312
                       1,                                          // tile1 = 1
                       (int)(AFRAG_BYTES),                         // stride0
                       0, 0 };
            __builtin_amdgcn_tensor_load_to_lds(g0, g1, gz4, gz4, gz8, 0);
        }

        // ---- descriptor 2: input patch (12 rows x 2176 B) ----
        {
            unsigned lds_off = (unsigned)(unsigned long long)(void*)&Vtile[0];
            unsigned long long ga = (unsigned long long)(const void*)V
                                  + (unsigned long long)i0 * ROW_B
                                  + (unsigned long long)j0 * REC_B;
            v4u g0 = { 1u, lds_off,
                       (unsigned)(ga & 0xFFFFFFFFull),
                       (unsigned)((ga >> 32) & 0x1FFFFFFull) | (2u << 30) };
            v8i g1 = { 0,                                          // data_size=1B
                       (int)((ROW_B & 0xFFFFu) << 16),             // dim0[15:0]
                       (int)(((ROW_B >> 16) & 0xFFFFu) |
                             ((unsigned)IMG_H << 16)),             // dim0 hi | dim1
                       (int)((STRIP_ROW_B & 0xFFFFu) << 16),       // tile0 = 2176
                       (int)(TILE_ROWS & 0xFFFFu),                 // tile1 = 12
                       (int)(ROW_B),                               // stride0 = 65536
                       0, 0 };
            __builtin_amdgcn_tensor_load_to_lds(g0, g1, gz4, gz4, gz8, 0);
        }
        __builtin_amdgcn_s_wait_tensorcnt(0);
    }
    __syncthreads();

    // -------- tile coordinates (wave-uniform) ------------------------------
    const int i = i0 + wave;              // this wave's output row
    if (i >= H_OUT) return;

    const int n  = lane & 15;             // N column within a tile
    const int th = lane >> 4;             // which tap of the K-pair

    v8f c[NTILES];
#pragma unroll
    for (int tt = 0; tt < NTILES; ++tt) c[tt] = (v8f){};

    // -------- K loop: 13 steps; A hoisted across the 4 j-tiles -------------
#pragma unroll
    for (int t = 0; t < 13; ++t) {
        v16h a = Alds[t][lane];

        int tap = 2 * t + th;             // 0..25 (25 = zero pad)
        int dy = tap / KW;
        int dx = tap - dy * KW;
        bool real = (tap < 25);

#pragma unroll
        for (int tt = 0; tt < NTILES; ++tt) {
            v16h b = {};
            if (real) {
                const v16h* p = (const v16h*)
                    &Vtile[(wave + dy) * STRIP_ROW_H + (tt * 16 + n + dx) * 16];
                b = *p;                   // 32 B: 2x ds_load_b128
            }
            c[tt] = __builtin_amdgcn_wmma_f32_16x16x32_f16(
                        false, a, false, b, (short)0, c[tt], false, false);
        }
    }

    // -------- store: lane (r, th) -> d = r + 8*th, column j0+tt*16+n -------
    const float bs    = bias[0];
    const int   dbase = th * 8;
#pragma unroll
    for (int tt = 0; tt < NTILES; ++tt) {
        const int j = j0 + tt * 16 + n;
        if (j < W_OUT) {
#pragma unroll
            for (int r = 0; r < 8; ++r) {
                int d = dbase + r;
                if (d < D_OUT) {
                    out[(size_t)d * H_OUT * W_OUT + (size_t)i * W_OUT + j]
                        = c[tt][r] + bs;
                }
            }
        }
    }
}

// ---------------------------------------------------------------------------
extern "C" void kernel_launch(void* const* d_in, const int* in_sizes, int n_in,
                              void* d_out, int out_size, void* d_ws, size_t ws_size,
                              hipStream_t stream) {
    const int*   x    = (const int*)d_in[0];
    const float* w    = (const float*)d_in[1];
    const float* bias = (const float*)d_in[2];
    float*       out  = (float*)d_out;

    // Workspace: [0, 128 MiB)  z-interleaved f16 bit-plane volume (L2-resident)
    //            [128 MiB, +13312 B)  WMMA A-fragment table
    _Float16* V    = (_Float16*)d_ws;
    v16h*     Atab = (v16h*)((char*)d_ws + V_BYTES);

    {
        int total  = IMG_H * IMG_W;
        int blocks = (total + 255) / 256;
        bitexpand_kernel<<<blocks, 256, 0, stream>>>(x, V);
    }
    {
        build_afrag_kernel<<<2, 256, 0, stream>>>(w, Atab);
    }
    {
        dim3 grid(IMG_W / (NTILES * 16), (H_OUT + 7) / 8); // 32 x 256 blocks
        conv_wmma_kernel<<<grid, 256, 0, stream>>>(V, Atab, bias, out);
    }
}
